// LSTM_43207370997853
// MI455X (gfx1250) — matrix operand order repaired and checked by reading, not compiled, not run on hardware
//
#include <hip/hip_runtime.h>

// ---------------- problem constants ----------------
constexpr int NB   = 2048;   // batch
constexpr int HID  = 1024;   // hidden
constexpr int G4   = 4096;   // 4*HID
constexpr int SEQN = 16;
constexpr int PF   = 28;     // pose features
constexpr int MF   = 14;     // mask features
constexpr int DEC  = 14;     // decoder steps
constexpr int KP   = 32;     // padded small-input K

typedef unsigned short u16;
typedef __attribute__((ext_vector_type(16))) __bf16 v16bf;
typedef __attribute__((ext_vector_type(8)))  float  v8f;

union Frag {
    v16bf v;
    struct { uint4 lo; uint4 hi; } q;
};

__device__ __forceinline__ u16 f2bf(float f) {
    union { float f; unsigned u; } x; x.f = f;
    unsigned u = x.u;
    u += 0x7FFFu + ((u >> 16) & 1u);   // round-to-nearest-even
    return (u16)(u >> 16);
}

__device__ __forceinline__ float sigm(float x) { return 1.0f / (1.0f + expf(-x)); }

// ---------------- setup kernels ----------------
__global__ void zero_f32(float* __restrict__ p, int n) {
    int i = blockIdx.x * blockDim.x + threadIdx.x;
    if (i < n) p[i] = 0.0f;
}
__global__ void zero_u16(u16* __restrict__ p, int n) {
    int i = blockIdx.x * blockDim.x + threadIdx.x;
    if (i < n) p[i] = 0;
}
// fp32 [4096, kin] -> bf16 [4096, kpad], zero padded
__global__ void conv_w(const float* __restrict__ W, u16* __restrict__ Wb,
                       int kin, int kpad, int total) {
    int i = blockIdx.x * blockDim.x + threadIdx.x;
    if (i >= total) return;
    int k = i % kpad, r = i / kpad;
    Wb[i] = (k < kin) ? f2bf(W[(size_t)r * kin + k]) : (u16)0;
}
__global__ void bias_comb(const float* __restrict__ a, const float* __restrict__ b,
                          float* __restrict__ o) {
    int i = blockIdx.x * blockDim.x + threadIdx.x;
    if (i < G4) o[i] = a[i] + b[i];
}
// src [B, SEQ, F] fp32 -> dst [SEQ, B, 32] bf16 zero-padded
__global__ void conv_inputs(const float* __restrict__ src, u16* __restrict__ dst, int F) {
    int i = blockIdx.x * blockDim.x + threadIdx.x;
    if (i >= SEQN * NB * 32) return;
    int k = i & 31; int bt = i >> 5;
    int b = bt & (NB - 1); int t = bt >> 11;   // NB = 2048
    dst[i] = (k < F) ? f2bf(src[((size_t)b * SEQN + t) * F + k]) : (u16)0;
}

// ---------------- fused gate GEMM:  G = bias + A1*W1^T + A2*W2^T ----------------
// A1 [NB,K1] bf16 row-major (K1 mult of 32), W1 [4096,K1] bf16 row-major,
// A2 [NB,1024] bf16, W2 [4096,1024] bf16, bias [4096] f32, G [NB,4096] f32.
// Wave tile: 32(M) x 64(N): 2 A-fragments x 4 B-fragments -> 8 WMMAs / 6 loads.
#define WMMA_BF16(A, Bf, C) \
    __builtin_amdgcn_wmma_f32_16x16x32_bf16(false, (A), false, (Bf), (short)0, (C), false, false)

__device__ __forceinline__ void mm_pass(const u16* __restrict__ A, const u16* __restrict__ W,
                                        int K, int rowA, int c0, int half,
                                        v8f& a00, v8f& a01, v8f& a02, v8f& a03,
                                        v8f& a10, v8f& a11, v8f& a12, v8f& a13) {
    const u16* ap0 = A + (size_t)rowA * K;           // rows Mbase..Mbase+15
    const u16* ap1 = ap0 + (size_t)16 * K;           // rows Mbase+16..Mbase+31
    const u16* w0 = W + (size_t)c0 * K;
    const u16* w1 = w0 + (size_t)16 * K;
    const u16* w2 = w0 + (size_t)32 * K;
    const u16* w3 = w0 + (size_t)48 * K;
    const int ao = half * 16;   // A: lane-half holds K = ao..ao+15 contiguous
    const int bo = half * 16;   // B: lane-half holds K = bo..bo+15 contiguous
    for (int kc = 0; kc < K; kc += 32) {
        Frag fa0, fa1, f0, f1, f2, f3;
        fa0.q.lo = *(const uint4*)(ap0 + kc + ao);
        fa0.q.hi = *(const uint4*)(ap0 + kc + ao + 8);
        fa1.q.lo = *(const uint4*)(ap1 + kc + ao);
        fa1.q.hi = *(const uint4*)(ap1 + kc + ao + 8);
        f0.q.lo = *(const uint4*)(w0 + kc + bo);
        f0.q.hi = *(const uint4*)(w0 + kc + bo + 8);
        f1.q.lo = *(const uint4*)(w1 + kc + bo);
        f1.q.hi = *(const uint4*)(w1 + kc + bo + 8);
        f2.q.lo = *(const uint4*)(w2 + kc + bo);
        f2.q.hi = *(const uint4*)(w2 + kc + bo + 8);
        f3.q.lo = *(const uint4*)(w3 + kc + bo);
        f3.q.hi = *(const uint4*)(w3 + kc + bo + 8);
        // reuse each B fragment across the two A row-fragments
        a00 = WMMA_BF16(fa0.v, f0.v, a00);
        a10 = WMMA_BF16(fa1.v, f0.v, a10);
        a01 = WMMA_BF16(fa0.v, f1.v, a01);
        a11 = WMMA_BF16(fa1.v, f1.v, a11);
        a02 = WMMA_BF16(fa0.v, f2.v, a02);
        a12 = WMMA_BF16(fa1.v, f2.v, a12);
        a03 = WMMA_BF16(fa0.v, f3.v, a03);
        a13 = WMMA_BF16(fa1.v, f3.v, a13);
    }
}

__global__ __launch_bounds__(256)
void gate_gemm(const u16* __restrict__ A1, const u16* __restrict__ W1, int K1,
               const u16* __restrict__ A2, const u16* __restrict__ W2,
               const float* __restrict__ bias, float* __restrict__ G) {
    const int lane = threadIdx.x & 31;
    const int w    = threadIdx.x >> 5;      // 0..7
    const int wm   = w >> 2, wn = w & 3;    // 2 x 4 wave grid
    const int half = lane >> 4, l = lane & 15;
    const int Mbase = blockIdx.y * 64 + wm * 32;
    const int Nbase = blockIdx.x * 256 + wn * 64;
    const int rowA  = Mbase + l;
    const int c0    = Nbase + l;

    v8f a00 = {}, a01 = {}, a02 = {}, a03 = {};
    v8f a10 = {}, a11 = {}, a12 = {}, a13 = {};
    mm_pass(A1, W1, K1,  rowA, c0, half, a00, a01, a02, a03, a10, a11, a12, a13);
    mm_pass(A2, W2, HID, rowA, c0, half, a00, a01, a02, a03, a10, a11, a12, a13);

    const float b0 = bias[c0], b1 = bias[c0 + 16], b2 = bias[c0 + 32], b3 = bias[c0 + 48];
    {
        float* grow = G + (size_t)(Mbase + half * 8) * G4;
#pragma unroll
        for (int r = 0; r < 8; ++r) {
            float* gr = grow + (size_t)r * G4;
            gr[c0]      = a00[r] + b0;
            gr[c0 + 16] = a01[r] + b1;
            gr[c0 + 32] = a02[r] + b2;
            gr[c0 + 48] = a03[r] + b3;
        }
    }
    {
        float* grow = G + (size_t)(Mbase + 16 + half * 8) * G4;
#pragma unroll
        for (int r = 0; r < 8; ++r) {
            float* gr = grow + (size_t)r * G4;
            gr[c0]      = a10[r] + b0;
            gr[c0 + 16] = a11[r] + b1;
            gr[c0 + 32] = a12[r] + b2;
            gr[c0 + 48] = a13[r] + b3;
        }
    }
}

// ---------------- pointwise LSTM cell update ----------------
__global__ __launch_bounds__(256)
void lstm_pointwise(const float* __restrict__ G, float* __restrict__ C,
                    float* __restrict__ Hst, u16* __restrict__ Hb) {
    int idx = blockIdx.x * blockDim.x + threadIdx.x;   // NB*HID threads
    int b = idx >> 10, j = idx & 1023;
    const float* g = G + (size_t)b * G4;
    float ig = g[j], fg = g[HID + j], gg = g[2 * HID + j], og = g[3 * HID + j];
    float c  = C[idx];
    float c2 = sigm(fg) * c + sigm(ig) * tanhf(gg);
    float h2 = sigm(og) * tanhf(c2);
    C[idx]   = c2;
    Hst[idx] = h2;
    Hb[idx]  = f2bf(h2);
}

// Y += X (fp32); optional bf16 mirror of Y
__global__ __launch_bounds__(256)
void add_state(float* __restrict__ Y, const float* __restrict__ X, u16* __restrict__ Yb) {
    int i = blockIdx.x * blockDim.x + threadIdx.x;
    float v = Y[i] + X[i];
    Y[i] = v;
    if (Yb) Yb[i] = f2bf(v);
}

// ---------------- FC heads ----------------
__global__ __launch_bounds__(256)
void fc_vel_kernel(const float* __restrict__ h, const float* __restrict__ W,
                   const float* __restrict__ bias, float* __restrict__ dout,
                   u16* __restrict__ xnext, int step) {
    int idx = blockIdx.x * blockDim.x + threadIdx.x;   // NB*32
    int b = idx >> 5, j = idx & 31;
    if (b >= NB) return;
    if (j >= PF) { xnext[b * 32 + j] = 0; return; }
    const float* hr = h + (size_t)b * HID;
    const float* wr = W + (size_t)j * HID;
    float s = bias[j];
    for (int k = 0; k < HID; ++k) s = fmaf(hr[k], wr[k], s);
    s = fminf(1.0f, fmaxf(-1.0f, s));
    dout[((size_t)b * DEC + step) * PF + j] = s;
    xnext[b * 32 + j] = f2bf(s);
}

__global__ __launch_bounds__(256)
void fc_mask_kernel(const float* __restrict__ h, const float* __restrict__ W,
                    const float* __restrict__ bias, float* __restrict__ dout,
                    u16* __restrict__ mbuf, int step) {
    int idx = blockIdx.x * blockDim.x + threadIdx.x;   // NB*16
    int b = idx >> 4, j = idx & 15;
    if (b >= NB) return;
    if (j >= MF) { mbuf[b * 32 + j] = 0; return; }
    const float* hr = h + (size_t)b * HID;
    const float* wr = W + (size_t)j * HID;
    float s = bias[j];
    for (int k = 0; k < HID; ++k) s = fmaf(hr[k], wr[k], s);
    s = sigm(s);
    dout[((size_t)b * DEC + step) * MF + j] = s;
    mbuf[b * 32 + j] = f2bf(s);
}

// ---------------- host orchestration ----------------
extern "C" void kernel_launch(void* const* d_in, const int* in_sizes, int n_in,
                              void* d_out, int out_size, void* d_ws, size_t ws_size,
                              hipStream_t stream) {
    (void)in_sizes; (void)n_in; (void)out_size; (void)ws_size;

    const float* pose = (const float*)d_in[0];
    const float* vel  = (const float*)d_in[1];
    const float* mask = (const float*)d_in[2];
    // cells in dict order: pose1, pose2, vel1, vel2, menc, mdec; each Wih, Whh, bih, bhh
    const float *cWih[6], *cWhh[6], *cBih[6], *cBhh[6];
    for (int c = 0; c < 6; ++c) {
        cWih[c] = (const float*)d_in[3 + 4 * c];
        cWhh[c] = (const float*)d_in[4 + 4 * c];
        cBih[c] = (const float*)d_in[5 + 4 * c];
        cBhh[c] = (const float*)d_in[6 + 4 * c];
    }
    const float* fcvW = (const float*)d_in[27];
    const float* fcvB = (const float*)d_in[28];
    const float* fcmW = (const float*)d_in[29];
    const float* fcmB = (const float*)d_in[30];

    float* dvel  = (float*)d_out;
    float* dmask = dvel + (size_t)NB * DEC * PF;

    // ---- workspace bump allocator ----
    char* wsb = (char*)d_ws;
    size_t off = 0;
    auto take = [&](size_t bytes) -> void* {
        void* p = wsb + off;
        off += (bytes + 255) & ~(size_t)255;
        return p;
    };

    const int kin[6]  = {PF, HID, PF, HID, MF, MF};
    const int kpad[6] = {KP, HID, KP, HID, KP, KP};
    u16* wih[6]; u16* whh[6]; float* bsum[6];
    for (int c = 0; c < 6; ++c) {
        wih[c]  = (u16*)take((size_t)G4 * kpad[c] * 2);
        whh[c]  = (u16*)take((size_t)G4 * HID * 2);
        bsum[c] = (float*)take((size_t)G4 * 4);
    }
    const size_t SH = (size_t)NB * HID;
    float* st = (float*)take(10 * SH * 4);
    float *hp1 = st,          *cp1 = st + SH,     *hp2 = st + 2 * SH, *cp2 = st + 3 * SH;
    float *hv1 = st + 4 * SH, *cv1 = st + 5 * SH, *hv2 = st + 6 * SH, *cv2 = st + 7 * SH;
    float *hd  = st + 8 * SH, *cd  = st + 9 * SH;

    const size_t BFN = 5 * SH + 2 * (size_t)NB * 32;
    u16* bb = (u16*)take(BFN * 2);
    u16 *hp1b = bb,          *hp2b = bb + SH,     *hv1b = bb + 2 * SH;
    u16 *hv2b = bb + 3 * SH, *hdb  = bb + 4 * SH;
    u16 *xnextb = bb + 5 * SH;
    u16 *mbuf   = xnextb + (size_t)NB * 32;

    float* Gbuf = (float*)take((size_t)NB * G4 * 4);
    u16* poseb = (u16*)take((size_t)SEQN * NB * 32 * 2);
    u16* velb  = (u16*)take((size_t)SEQN * NB * 32 * 2);
    u16* maskb = (u16*)take((size_t)SEQN * NB * 32 * 2);

    // ---- init & conversions ----
    {
        int n = (int)(10 * SH);
        zero_f32<<<(n + 255) / 256, 256, 0, stream>>>(st, n);
        int m = (int)BFN;
        zero_u16<<<(m + 255) / 256, 256, 0, stream>>>(bb, m);
    }
    for (int c = 0; c < 6; ++c) {
        int t1 = G4 * kpad[c];
        conv_w<<<(t1 + 255) / 256, 256, 0, stream>>>(cWih[c], wih[c], kin[c], kpad[c], t1);
        int t2 = G4 * HID;
        conv_w<<<(t2 + 255) / 256, 256, 0, stream>>>(cWhh[c], whh[c], HID, HID, t2);
        bias_comb<<<G4 / 256, 256, 0, stream>>>(cBih[c], cBhh[c], bsum[c]);
    }
    {
        int n = SEQN * NB * 32;
        conv_inputs<<<(n + 255) / 256, 256, 0, stream>>>(pose, poseb, PF);
        conv_inputs<<<(n + 255) / 256, 256, 0, stream>>>(vel,  velb,  PF);
        conv_inputs<<<(n + 255) / 256, 256, 0, stream>>>(mask, maskb, MF);
    }

    dim3 ggrid(G4 / 256, NB / 64);                 // (16, 32): 64M x 256N block tiles
    const int pwBlocks = (int)(SH / 256);          // 8192
    auto GATES = [&](const u16* A1, int c, const u16* Hb) {
        gate_gemm<<<ggrid, 256, 0, stream>>>(A1, wih[c], kpad[c], Hb, whh[c], bsum[c], Gbuf);
    };
    auto PW = [&](float* C, float* Hst, u16* Hb) {
        lstm_pointwise<<<pwBlocks, 256, 0, stream>>>(Gbuf, C, Hst, Hb);
    };

    // ---- encoder: 15 steps (pose_t[1:], vel_t[:-1]) ----
    for (int t = 0; t < SEQN - 1; ++t) {
        GATES(poseb + (size_t)(t + 1) * NB * 32, 0, hp1b); PW(cp1, hp1, hp1b);
        GATES(hp1b,                              1, hp2b); PW(cp2, hp2, hp2b);
        GATES(velb + (size_t)t * NB * 32,        2, hv1b); PW(cv1, hv1, hv1b);
        GATES(hv1b,                              3, hv2b); PW(cv2, hv2, hv2b);
    }
    // residuals: hv1 += hp1 ; hv2 += hp2
    add_state<<<pwBlocks, 256, 0, stream>>>(hv1, hp1, hv1b);
    add_state<<<pwBlocks, 256, 0, stream>>>(hv2, hp2, hv2b);

    // ---- mask encoder: 16 steps ----
    for (int t = 0; t < SEQN; ++t) {
        GATES(maskb + (size_t)t * NB * 32, 4, hdb);
        PW(cd, hd, hdb);
    }

    // ---- decoder: 14 steps ----
    for (int s = 0; s < DEC; ++s) {
        const u16* xin = (s == 0) ? (velb + (size_t)(SEQN - 1) * NB * 32) : xnextb;
        GATES(xin,  2, hv1b); PW(cv1, hv1, hv1b);
        GATES(hv1b, 3, hv2b); PW(cv2, hv2, hv2b);
        fc_vel_kernel<<<(NB * 32) / 256, 256, 0, stream>>>(hv2, fcvW, fcvB, dvel, xnextb, s);
        add_state<<<pwBlocks, 256, 0, stream>>>(hd, hv2, hdb);
        add_state<<<pwBlocks, 256, 0, stream>>>(cd, cv2, (u16*)nullptr);
        const u16* min_ = (s == 0) ? (maskb + (size_t)(SEQN - 1) * NB * 32) : mbuf;
        GATES(min_, 5, hdb); PW(cd, hd, hdb);
        fc_mask_kernel<<<(NB * 16) / 256, 256, 0, stream>>>(hd, fcmW, fcmB, dmask, mbuf, s);
    }
}
